// MambaBlock_16011638079758
// MI455X (gfx1250) — matrix operand order repaired and checked by reading, not compiled, not run on hardware
//
#include <hip/hip_runtime.h>
#include <hip/hip_bf16.h>
#include <math.h>

// ---------------------------------------------------------------------------
// MambaBlock for MI455X (gfx1250), wave32, WMMA bf16 path.
// B=8, T=4096, D=1024, N=16, M=B*T=32768.
// Pipeline:
//   1) transpose_bf16: Wd,Wo,Wb,Wc (fp32 row-major [K][N]) -> bf16 [N][K]
//   2) sa_kernel: sA[n] = sigmoid(mean_d A[d][n])
//   3) proj_kernel: per 32-row tile, WMMA x@Wd -> softplus -> row-mean (dbar),
//      plus WMMA x@Wb -> Bt and x@Wc -> Ct (delta never hits HBM)
//   4) scan_kernel: chunked parallel linear recurrence, emits s_t = sum_n h*C
//   5) out_kernel: WMMA (s .* x) @ Wo + bo -> d_out
// ---------------------------------------------------------------------------

typedef __bf16 bf16_t;
typedef __attribute__((ext_vector_type(16))) __bf16 v16bf;
typedef __attribute__((ext_vector_type(8)))  __bf16 v8bf;
typedef __attribute__((ext_vector_type(4)))  __bf16 v4bf;
typedef __attribute__((ext_vector_type(8)))  float  v8f;

#define BB    8
#define TT    4096
#define DD    1024
#define NN    16
#define MM    (BB * TT)        // 32768 rows

// ---------------------------------------------------------------------------
// Weight transpose + fp32 -> bf16 convert. src[rows][cols] -> dst[cols][rows]
// ---------------------------------------------------------------------------
__global__ __launch_bounds__(256)
void transpose_bf16(const float* __restrict__ src, bf16_t* __restrict__ dst,
                    int rows, int cols) {
  __shared__ float tile[32][33];
  const int c0 = blockIdx.x * 32, r0 = blockIdx.y * 32;
  const int tx = threadIdx.x, ty = threadIdx.y;       // block (32, 8)
  for (int j = ty; j < 32; j += 8) {
    int r = r0 + j, c = c0 + tx;
    if (r < rows && c < cols) tile[j][tx] = src[r * cols + c];
  }
  __syncthreads();
  for (int j = ty; j < 32; j += 8) {
    int c = c0 + j, r = r0 + tx;
    if (c < cols && r < rows) dst[c * rows + r] = (bf16_t)tile[tx][j];
  }
}

// ---------------------------------------------------------------------------
// sA[n] = sigmoid(mean over D of A[:, n])
// ---------------------------------------------------------------------------
__global__ void sa_kernel(const float* __restrict__ A, float* __restrict__ sA) {
  int n = threadIdx.x;
  if (n < NN) {
    float sum = 0.f;
    for (int d = 0; d < DD; ++d) sum += A[d * NN + n];
    float m = sum * (1.f / (float)DD);
    sA[n] = 1.f / (1.f + expf(-m));
  }
}

// ---------------------------------------------------------------------------
// WMMA fragment helpers (layouts per CDNA5 ISA 7.12.2)
// ---------------------------------------------------------------------------
__device__ inline v16bf load_a_frag(const bf16_t* __restrict__ row, int k0, int half) {
  // A 16-bit 16x32: lane-half owns K chunks [k0+8h, +8) and [k0+16+8h, +8)
  v8bf a0 = *(const v8bf*)(row + k0 + 8 * half);
  v8bf a1 = *(const v8bf*)(row + k0 + 16 + 8 * half);
  v16bf a;
#pragma unroll
  for (int q = 0; q < 8; ++q) { a[q] = a0[q]; a[8 + q] = a1[q]; }
  return a;
}

__device__ inline v16bf load_b_frag(const bf16_t* __restrict__ colT, int k0, int half) {
  // B 16-bit 32x16: lane = column, K chunks [k0+8h, +8) and [k0+16+8h, +8)
  v8bf b0 = *(const v8bf*)(colT + k0 + 8 * half);
  v8bf b1 = *(const v8bf*)(colT + k0 + 16 + 8 * half);
  v16bf b;
#pragma unroll
  for (int q = 0; q < 8; ++q) { b[q] = b0[q]; b[8 + q] = b1[q]; }
  return b;
}

__device__ inline float softplus_f(float v) {
  return (v > 20.f) ? v : log1pf(expf(v));
}

// ---------------------------------------------------------------------------
// proj_kernel: 32 rows per workgroup, 8 waves.
//   wave -> (rowsub = w>>2 in {0,1}, colgroup = w&3).
//   Each (rowsub, colgroup) wave: 16 column tiles of Wd (covering 1024 cols),
//   K-loop of 32 per tile (K=1024). Epilogue: softplus, accumulate row sums
//   via LDS ds_add_f32. colgroup 0/1 additionally compute Bt / Ct.
// ---------------------------------------------------------------------------
__global__ __launch_bounds__(256)
void proj_kernel(const float* __restrict__ x,
                 const bf16_t* __restrict__ WdT, const float* __restrict__ bd,
                 const bf16_t* __restrict__ WbT, const float* __restrict__ bb,
                 const bf16_t* __restrict__ WcT, const float* __restrict__ bc,
                 float* __restrict__ dbar, float* __restrict__ Bt,
                 float* __restrict__ Ct) {
  __shared__ bf16_t xA[32 * DD];        // 64 KB
  __shared__ float rowsum[32];
  const int tid = threadIdx.x;
  const int rowbase = blockIdx.x * 32;

  if (tid < 32) rowsum[tid] = 0.f;

  // Stage x tile (fp32 -> bf16) into LDS, coalesced float4 loads.
  const float4* xs = (const float4*)(x + (size_t)rowbase * DD);
  for (int i = tid; i < 32 * DD / 4; i += 256) {
    float4 f = xs[i];
    v4bf p;
    p[0] = (bf16_t)f.x; p[1] = (bf16_t)f.y; p[2] = (bf16_t)f.z; p[3] = (bf16_t)f.w;
    *(v4bf*)&xA[i * 4] = p;
  }
  __syncthreads();

  const int wave = tid >> 5, lane = tid & 31;
  const int rs = wave >> 2, cg = wave & 3;
  const int half = lane >> 4, ln = lane & 15;
  const bf16_t* arow = &xA[(rs * 16 + ln) * DD];

  float spsum[8];
#pragma unroll
  for (int r = 0; r < 8; ++r) spsum[r] = 0.f;

  // Main GEMM: x @ Wd, softplus + row reduction, no delta materialization.
  for (int it = 0; it < 16; ++it) {
    const int colb = (cg + it * 4) * 16;
    const bf16_t* bcol = WdT + (size_t)(colb + ln) * DD;
    v8f acc = {};
    for (int k0 = 0; k0 < DD; k0 += 32) {
      v16bf a = load_a_frag(arow, k0, half);
      v16bf b = load_b_frag(bcol, k0, half);
      acc = __builtin_amdgcn_wmma_f32_16x16x32_bf16(
          false, a, false, b, (short)0, acc, false, false);
    }
    const float bias = bd[colb + ln];
#pragma unroll
    for (int r = 0; r < 8; ++r) spsum[r] += softplus_f(acc[r] + bias);
  }
#pragma unroll
  for (int r = 0; r < 8; ++r)
    atomicAdd(&rowsum[rs * 16 + r + 8 * half], spsum[r]);   // ds_add_f32

  // Skinny GEMMs: colgroup 0 -> Bt = x@Wb + bb, colgroup 1 -> Ct = x@Wc + bc.
  if (cg < 2) {
    const bf16_t* WT = (cg == 0) ? WbT : WcT;
    const float*  bv = (cg == 0) ? bb : bc;
    float*       out = (cg == 0) ? Bt : Ct;
    const bf16_t* bcol = WT + (size_t)ln * DD;
    v8f acc = {};
    for (int k0 = 0; k0 < DD; k0 += 32) {
      v16bf a = load_a_frag(arow, k0, half);
      v16bf b = load_b_frag(bcol, k0, half);
      acc = __builtin_amdgcn_wmma_f32_16x16x32_bf16(
          false, a, false, b, (short)0, acc, false, false);
    }
    const float bias = bv[ln];
#pragma unroll
    for (int r = 0; r < 8; ++r) {
      int rowg = rowbase + rs * 16 + r + 8 * half;
      out[(size_t)rowg * NN + ln] = acc[r] + bias;
    }
  }

  __syncthreads();
  if (tid < 32) dbar[rowbase + tid] = rowsum[tid] * (1.f / (float)DD);
}

// ---------------------------------------------------------------------------
// scan_kernel: one block (1024 thr) per batch b. h_t = a_t h_{t-1} + b_t with
// a_t = 1 - dbar_t (shared over n), b_t[n] = dbar_t * Bt[t][n] * sA[n].
// Chunked scan: 64 chunks x 64 steps; emits s_t = sum_n h_t[n] * Ct[t][n].
// ---------------------------------------------------------------------------
__global__ __launch_bounds__(1024)
void scan_kernel(const float* __restrict__ dbar, const float* __restrict__ Bt,
                 const float* __restrict__ Ct, const float* __restrict__ sA,
                 float* __restrict__ s) {
  __shared__ float P[64];          // chunk decay products (n-independent)
  __shared__ float Q[NN * 64];     // chunk composed offsets
  __shared__ float Cy[NN * 64];    // carry-in h at chunk starts
  const int b = blockIdx.x, tid = threadIdx.x;
  const int n = tid & 15, c = tid >> 4;        // c in [0, 64)
  const int base = b * TT;
  const float sa = sA[n];

  // Phase 1: per-(n, chunk) composition over 64 steps.
  {
    float Pp = 1.f, Qq = 0.f;
    const int t0 = c * 64;
    for (int t = t0; t < t0 + 64; ++t) {
      float db = dbar[base + t];
      float a  = 1.f - db;
      float bt = db * Bt[(size_t)(base + t) * NN + n] * sa;
      Qq = a * Qq + bt;
      Pp *= a;
    }
    Q[n * 64 + c] = Qq;
    if (n == 0) P[c] = Pp;
  }
  __syncthreads();

  // Phase 2: scan 64 chunk summaries per n (16 lanes).
  if (tid < NN) {
    float H = 0.f;
    for (int cc = 0; cc < 64; ++cc) {
      Cy[tid * 64 + cc] = H;
      H = P[cc] * H + Q[tid * 64 + cc];
    }
  }
  __syncthreads();

  // Phase 3: 64 lanes, one chunk each; full 16-state recurrence in registers,
  // emitting the scalar projection s_t directly.
  if (tid < 64) {
    float h[NN], sal[NN];
#pragma unroll
    for (int j = 0; j < NN; ++j) { h[j] = Cy[j * 64 + tid]; sal[j] = sA[j]; }
    const int t0 = tid * 64;
    for (int t = t0; t < t0 + 64; ++t) {
      float db = dbar[base + t];
      float a  = 1.f - db;
      const float4* bt4 = (const float4*)(Bt + (size_t)(base + t) * NN);
      const float4* ct4 = (const float4*)(Ct + (size_t)(base + t) * NN);
      float acc = 0.f;
#pragma unroll
      for (int q = 0; q < 4; ++q) {
        float4 bv = bt4[q], cv = ct4[q];
        h[4*q+0] = a * h[4*q+0] + db * bv.x * sal[4*q+0]; acc += h[4*q+0] * cv.x;
        h[4*q+1] = a * h[4*q+1] + db * bv.y * sal[4*q+1]; acc += h[4*q+1] * cv.y;
        h[4*q+2] = a * h[4*q+2] + db * bv.z * sal[4*q+2]; acc += h[4*q+2] * cv.z;
        h[4*q+3] = a * h[4*q+3] + db * bv.w * sal[4*q+3]; acc += h[4*q+3] * cv.w;
      }
      s[base + t] = acc;
    }
  }
}

// ---------------------------------------------------------------------------
// out_kernel: out = (s .* x) @ Wo + bo. Scale fused into the bf16 staging.
// ---------------------------------------------------------------------------
__global__ __launch_bounds__(256)
void out_kernel(const float* __restrict__ x, const float* __restrict__ s,
                const bf16_t* __restrict__ WoT, const float* __restrict__ bo,
                float* __restrict__ out) {
  __shared__ bf16_t xA[32 * DD];        // 64 KB
  const int tid = threadIdx.x;
  const int rowbase = blockIdx.x * 32;

  const float4* xs = (const float4*)(x + (size_t)rowbase * DD);
  for (int i = tid; i < 32 * DD / 4; i += 256) {
    float sv = s[rowbase + (i >> 8)];   // row = (i*4) / 1024
    float4 f = xs[i];
    v4bf p;
    p[0] = (bf16_t)(f.x * sv); p[1] = (bf16_t)(f.y * sv);
    p[2] = (bf16_t)(f.z * sv); p[3] = (bf16_t)(f.w * sv);
    *(v4bf*)&xA[i * 4] = p;
  }
  __syncthreads();

  const int wave = tid >> 5, lane = tid & 31;
  const int rs = wave >> 2, cg = wave & 3;
  const int half = lane >> 4, ln = lane & 15;
  const bf16_t* arow = &xA[(rs * 16 + ln) * DD];

  for (int it = 0; it < 16; ++it) {
    const int colb = (cg + it * 4) * 16;
    const bf16_t* bcol = WoT + (size_t)(colb + ln) * DD;
    v8f acc = {};
    for (int k0 = 0; k0 < DD; k0 += 32) {
      v16bf a = load_a_frag(arow, k0, half);
      v16bf b = load_b_frag(bcol, k0, half);
      acc = __builtin_amdgcn_wmma_f32_16x16x32_bf16(
          false, a, false, b, (short)0, acc, false, false);
    }
    const float bias = bo[colb + ln];
#pragma unroll
    for (int r = 0; r < 8; ++r) {
      int rowg = rowbase + rs * 16 + r + 8 * half;
      out[(size_t)rowg * DD + colb + ln] = acc[r] + bias;
    }
  }
}

// ---------------------------------------------------------------------------
// Host launcher
// ---------------------------------------------------------------------------
extern "C" void kernel_launch(void* const* d_in, const int* in_sizes, int n_in,
                              void* d_out, int out_size, void* d_ws, size_t ws_size,
                              hipStream_t stream) {
  const float* x  = (const float*)d_in[0];
  const float* Wd = (const float*)d_in[1];
  const float* bd = (const float*)d_in[2];
  const float* Wb = (const float*)d_in[3];
  const float* bb = (const float*)d_in[4];
  const float* Wc = (const float*)d_in[5];
  const float* bc = (const float*)d_in[6];
  const float* Wo = (const float*)d_in[7];
  const float* bo = (const float*)d_in[8];
  const float* A  = (const float*)d_in[9];
  float* out = (float*)d_out;

  // Workspace carve-up (~8.5 MB total).
  char* w = (char*)d_ws;
  bf16_t* WdT = (bf16_t*)w;  w += (size_t)DD * DD * sizeof(bf16_t);   // 2 MB
  bf16_t* WoT = (bf16_t*)w;  w += (size_t)DD * DD * sizeof(bf16_t);   // 2 MB
  bf16_t* WbT = (bf16_t*)w;  w += (size_t)NN * DD * sizeof(bf16_t);   // 32 KB
  bf16_t* WcT = (bf16_t*)w;  w += (size_t)NN * DD * sizeof(bf16_t);   // 32 KB
  float*  sA  = (float*)w;   w += 256;
  float*  dbar= (float*)w;   w += (size_t)MM * sizeof(float);          // 128 KB
  float*  BtA = (float*)w;   w += (size_t)MM * NN * sizeof(float);     // 2 MB
  float*  CtA = (float*)w;   w += (size_t)MM * NN * sizeof(float);     // 2 MB
  float*  sArr= (float*)w;   w += (size_t)MM * sizeof(float);          // 128 KB
  (void)ws_size; (void)in_sizes; (void)n_in; (void)out_size;

  dim3 tblk(32, 8);
  // Wd, Wo: 1024x1024 -> transposed bf16.
  transpose_bf16<<<dim3(DD / 32, DD / 32), tblk, 0, stream>>>(Wd, WdT, DD, DD);
  transpose_bf16<<<dim3(DD / 32, DD / 32), tblk, 0, stream>>>(Wo, WoT, DD, DD);
  // Wb, Wc: 1024x16 -> 16x1024 bf16.
  transpose_bf16<<<dim3(1, DD / 32), tblk, 0, stream>>>(Wb, WbT, DD, NN);
  transpose_bf16<<<dim3(1, DD / 32), tblk, 0, stream>>>(Wc, WcT, DD, NN);

  sa_kernel<<<1, 32, 0, stream>>>(A, sA);

  proj_kernel<<<MM / 32, 256, 0, stream>>>(x, WdT, bd, WbT, bb, WcT, bc,
                                           dbar, BtA, CtA);

  scan_kernel<<<BB, 1024, 0, stream>>>(dbar, BtA, CtA, sA, sArr);

  out_kernel<<<MM / 32, 256, 0, stream>>>(x, sArr, WoT, bo, out);
}
